// Gemma3p5DecoderLayer_39771397161179
// MI455X (gfx1250) — matrix operand order repaired and checked by reading, not compiled, not run on hardware
//
#include <hip/hip_runtime.h>
#include <hip/hip_bf16.h>

typedef __attribute__((ext_vector_type(16))) _Float16 v16h;
typedef __attribute__((ext_vector_type(8)))  _Float16 v8h;
typedef __attribute__((ext_vector_type(4)))  _Float16 h4;
typedef __attribute__((ext_vector_type(8)))  float    v8f;

#define DD        2048
#define SEQ       2048
#define BB        2
#define BS_       4096
#define NH        8
#define NKV       2
#define HD        256
#define FFDIM     8192
#define NIM       4
#define EPSF      1e-6f
#define QSC       0.0625f          /* 256^-0.5 */
#define INV_SQRT2 0.70710678118654752f

// gfx1250 async global->LDS copy (ASYNCcnt-tracked). Flip to 0 if the
// assembler rejects the mnemonic.
#define USE_ASYNC_LDS 1

union AFQ { v16h v; v8h q[2]; _Float16 h[16]; };
union CFu { v8f v; float f[8]; };
union H4u { h4 v; _Float16 h[4]; };

__device__ __forceinline__ v8f wmma_f16(v16h a, v16h b, v8f c) {
  return __builtin_amdgcn_wmma_f32_16x16x32_f16(false, a, false, b, (short)0, c,
                                                false, false);
}

#if USE_ASYNC_LDS
// 16B async copy: global -> LDS, tracked by ASYNCcnt. vdst = per-lane LDS
// byte offset (low 32 bits of the generic pointer), vaddr = 64-bit global.
__device__ __forceinline__ void async_copy16(const _Float16* g, _Float16* l) {
  unsigned int loff = (unsigned int)(size_t)l;
  asm volatile("global_load_async_to_lds_b128 %0, %1, off"
               :
               : "v"(loff), "v"(g)
               : "memory");
}
__device__ __forceinline__ void wait_async0() {
  asm volatile("s_wait_asynccnt 0x0" ::: "memory");
}
#endif

__device__ __forceinline__ float clipw(float w) {
  return fminf(fmaxf(w, -120.0f), 120.0f);
}

// exact rewrite of 0.5x(1+tanh(z)) as logistic; branch-free
__device__ __forceinline__ float gelu_tanh(float x) {
  float x3 = x * x * x;
  float z2 = 1.5957691216057308f * (x + 0.044715f * x3);  // 2*sqrt(2/pi)*(..)
  return x / (1.0f + __expf(-z2));
}

__device__ __forceinline__ float tanh_fast(float y) {
  return 1.0f - 2.0f / (__expf(2.0f * y) + 1.0f);
}

// blockDim.x == 256 (8 wave32)
__device__ __forceinline__ float block_reduce_sum(float v, float* sbuf) {
  const int lane = threadIdx.x & 31;
  const int w = threadIdx.x >> 5;
#pragma unroll
  for (int o = 16; o >= 1; o >>= 1) v += __shfl_xor(v, o, 32);
  if (lane == 0) sbuf[w] = v;
  __syncthreads();
  if (w == 0) {
    float r = (lane < 8) ? sbuf[lane] : 0.0f;
#pragma unroll
    for (int o = 4; o >= 1; o >>= 1) r += __shfl_xor(r, o, 32);
    if (lane == 0) sbuf[0] = r;
  }
  __syncthreads();
  float out = sbuf[0];
  __syncthreads();
  return out;
}

// ---------------------------------------------------------------------------
__global__ __launch_bounds__(256) void modalities_kernel(
    const float* __restrict__ h, const float* __restrict__ rnw,
    const float* __restrict__ wm, float* __restrict__ mout) {
  __shared__ float sbuf[8];
  const size_t t = blockIdx.x;
  const float* row = h + t * (size_t)DD;
  float x[8];
  float ss = 0.f;
#pragma unroll
  for (int e = 0; e < 8; ++e) {
    x[e] = row[threadIdx.x + 256 * e];
    ss += x[e] * x[e];
  }
  ss = block_reduce_sum(ss, sbuf);
  const float r = rsqrtf(ss * (1.0f / DD) + EPSF);
  float dot[NIM] = {0.f, 0.f, 0.f, 0.f};
#pragma unroll
  for (int e = 0; e < 8; ++e) {
    const int d = threadIdx.x + 256 * e;
    const float hn = x[e] * r * rnw[d];
#pragma unroll
    for (int k = 0; k < NIM; ++k) dot[k] += hn * wm[(size_t)k * DD + d];
  }
  for (int k = 0; k < NIM; ++k) {
    float s = block_reduce_sum(dot[k], sbuf);
    if (threadIdx.x == 0) mout[t * NIM + k] = tanh_fast(s * (1.0f / DD));
  }
}

// ---------------------------------------------------------------------------
__global__ __launch_bounds__(256) void coefs_act_kernel(
    const float* __restrict__ X, const float* __restrict__ m,
    const float* __restrict__ pcw, const float* __restrict__ iln,
    float* __restrict__ coefs, float* __restrict__ act,
    float* __restrict__ actn) {
  __shared__ float sbuf[8];
  const size_t t = blockIdx.x;
  float m4[NIM];
#pragma unroll
  for (int k = 0; k < NIM; ++k) m4[k] = m[t * NIM + k];
  if (threadIdx.x < NIM * NIM) {
    float c = 0.f;
#pragma unroll
    for (int k = 0; k < NIM; ++k) c += m4[k] * clipw(pcw[threadIdx.x * NIM + k]);
    coefs[t * (NIM * NIM) + threadIdx.x] = c;
  }
  float c0[NIM];
#pragma unroll
  for (int j = 0; j < NIM; ++j) {
    float c = 0.f;
#pragma unroll
    for (int k = 0; k < NIM; ++k) c += m4[k] * clipw(pcw[j * NIM + k]);
    c0[j] = c;
  }
  const size_t plane = (size_t)BS_ * DD;
  float a[8];
  float ss = 0.f;
#pragma unroll
  for (int e = 0; e < 8; ++e) {
    const size_t idx = t * (size_t)DD + threadIdx.x + 256 * e;
    float v = X[idx];
#pragma unroll
    for (int j = 0; j < NIM; ++j) v += c0[j] * X[(size_t)j * plane + idx];
    a[e] = v;
    act[idx] = v;
    ss += v * v;
  }
  ss = block_reduce_sum(ss, sbuf);
  const float r = rsqrtf(ss * (1.0f / DD) + EPSF);
#pragma unroll
  for (int e = 0; e < 8; ++e) {
    const int d = threadIdx.x + 256 * e;
    const size_t idx = t * (size_t)DD + d;
    actn[idx] = a[e] * r * (iln[d] + 1.0f);
  }
}

// ---------------------------------------------------------------------------
// C[M,N] = A[M,K] * W[N,K]^T via f16 WMMA.  Block tile 128x64, k-step 32.
// A tile row-major in LDS; W tile stored K-MAJOR (Wt[k][n]) and B-fragments
// are gathered one half per k row.  8 waves = (mw 0..3) x (nw 0..1); each wave
// computes 32x32 = 4 WMMA tiles per k-step.
template <typename AT>
__global__ __launch_bounds__(256) void gemm_nt_kernel(
    const AT* __restrict__ A, const float* __restrict__ W,
    float* __restrict__ C, int M, int N, int K) {
  __shared__ __align__(16) _Float16 As[128][40];
  __shared__ __align__(16) _Float16 Wt[32][72];
  const int tid = threadIdx.x;
  const int lane = tid & 31;
  const int l16 = lane & 15;
  const int wvid = tid >> 5;
  const int mw = wvid >> 1, nw = wvid & 1;
  const int m0 = blockIdx.x * 128, n0 = blockIdx.y * 64;
  CFu acc[2][2] = {};
  const int kb = (lane < 16) ? 0 : 8;
  const int ar = mw * 32 + l16;
  const int sr = tid >> 3;            // 0..31
  const int skq = (tid & 7) * 4;      // A stage: 4 consecutive k
  const int wn = tid >> 2;            // 0..63
  const int wk8 = (tid & 3) * 8;      // W stage: 8 consecutive k
  for (int k0 = 0; k0 < K; k0 += 32) {
#pragma unroll
    for (int it = 0; it < 4; ++it) {
      const int r = sr + it * 32;
      H4u t4;
#pragma unroll
      for (int j = 0; j < 4; ++j)
        t4.h[j] = (_Float16)(float)A[(size_t)(m0 + r) * K + k0 + skq + j];
      *(h4*)&As[r][skq] = t4.v;
    }
#pragma unroll
    for (int j = 0; j < 8; ++j)
      Wt[wk8 + j][wn] = (_Float16)W[(size_t)(n0 + wn) * K + k0 + wk8 + j];
    if (k0 + 32 < K) {  // deep-prefetch next k-step into GL2
      __builtin_prefetch(&W[(size_t)(n0 + wn) * K + k0 + 32 + wk8], 0, 1);
      __builtin_prefetch(&A[(size_t)(m0 + sr) * K + k0 + 32 + skq], 0, 1);
    }
    __syncthreads();
    AFQ a0, a1, b0, b1;
    a0.q[0] = *(const v8h*)&As[ar][kb];
    a0.q[1] = *(const v8h*)&As[ar][kb + 16];
    a1.q[0] = *(const v8h*)&As[ar + 16][kb];
    a1.q[1] = *(const v8h*)&As[ar + 16][kb + 16];
#pragma unroll
    for (int s = 0; s < 16; ++s) {
      const int kk = kb + (s & 7) + ((s >> 3) << 4);
      b0.h[s] = Wt[kk][nw * 32 + l16];
      b1.h[s] = Wt[kk][nw * 32 + 16 + l16];
    }
    acc[0][0].v = wmma_f16(a0.v, b0.v, acc[0][0].v);
    acc[0][1].v = wmma_f16(a0.v, b1.v, acc[0][1].v);
    acc[1][0].v = wmma_f16(a1.v, b0.v, acc[1][0].v);
    acc[1][1].v = wmma_f16(a1.v, b1.v, acc[1][1].v);
    __syncthreads();
  }
  const int halfsel = (lane >> 4) * 8;
#pragma unroll
  for (int mi = 0; mi < 2; ++mi)
#pragma unroll
    for (int ni = 0; ni < 2; ++ni) {
      const int ccol = n0 + nw * 32 + ni * 16 + l16;
#pragma unroll
      for (int v2 = 0; v2 < 8; ++v2)
        C[(size_t)(m0 + mw * 32 + mi * 16 + halfsel + v2) * N + ccol] =
            acc[mi][ni].f[v2];
    }
}

// ---------------------------------------------------------------------------
// Fused gate+up GEMM: FF[M,N] = (f16) gelu(A*Wg^T) * (A*Wu^T)
__global__ __launch_bounds__(256) void gemm_gate_up_kernel(
    const float* __restrict__ A, const float* __restrict__ Wg,
    const float* __restrict__ Wu, _Float16* __restrict__ FF, int M, int N,
    int K) {
  __shared__ __align__(16) _Float16 As[128][40];
  __shared__ __align__(16) _Float16 Gt[32][72];
  __shared__ __align__(16) _Float16 Ut[32][72];
  const int tid = threadIdx.x;
  const int lane = tid & 31;
  const int l16 = lane & 15;
  const int wvid = tid >> 5;
  const int mw = wvid >> 1, nw = wvid & 1;
  const int m0 = blockIdx.x * 128, n0 = blockIdx.y * 64;
  CFu g[2][2] = {}, u[2][2] = {};
  const int kb = (lane < 16) ? 0 : 8;
  const int ar = mw * 32 + l16;
  const int sr = tid >> 3;
  const int skq = (tid & 7) * 4;
  const int wn = tid >> 2;
  const int wk8 = (tid & 3) * 8;
  for (int k0 = 0; k0 < K; k0 += 32) {
#pragma unroll
    for (int it = 0; it < 4; ++it) {
      const int r = sr + it * 32;
      H4u t4;
#pragma unroll
      for (int j = 0; j < 4; ++j)
        t4.h[j] = (_Float16)A[(size_t)(m0 + r) * K + k0 + skq + j];
      *(h4*)&As[r][skq] = t4.v;
    }
#pragma unroll
    for (int j = 0; j < 8; ++j) {
      Gt[wk8 + j][wn] = (_Float16)Wg[(size_t)(n0 + wn) * K + k0 + wk8 + j];
      Ut[wk8 + j][wn] = (_Float16)Wu[(size_t)(n0 + wn) * K + k0 + wk8 + j];
    }
    if (k0 + 32 < K) {
      __builtin_prefetch(&Wg[(size_t)(n0 + wn) * K + k0 + 32 + wk8], 0, 1);
      __builtin_prefetch(&Wu[(size_t)(n0 + wn) * K + k0 + 32 + wk8], 0, 1);
    }
    __syncthreads();
    AFQ a0, a1, bg0, bg1, bu0, bu1;
    a0.q[0] = *(const v8h*)&As[ar][kb];
    a0.q[1] = *(const v8h*)&As[ar][kb + 16];
    a1.q[0] = *(const v8h*)&As[ar + 16][kb];
    a1.q[1] = *(const v8h*)&As[ar + 16][kb + 16];
#pragma unroll
    for (int s = 0; s < 16; ++s) {
      const int kk = kb + (s & 7) + ((s >> 3) << 4);
      bg0.h[s] = Gt[kk][nw * 32 + l16];
      bg1.h[s] = Gt[kk][nw * 32 + 16 + l16];
      bu0.h[s] = Ut[kk][nw * 32 + l16];
      bu1.h[s] = Ut[kk][nw * 32 + 16 + l16];
    }
    g[0][0].v = wmma_f16(a0.v, bg0.v, g[0][0].v);
    g[0][1].v = wmma_f16(a0.v, bg1.v, g[0][1].v);
    g[1][0].v = wmma_f16(a1.v, bg0.v, g[1][0].v);
    g[1][1].v = wmma_f16(a1.v, bg1.v, g[1][1].v);
    u[0][0].v = wmma_f16(a0.v, bu0.v, u[0][0].v);
    u[0][1].v = wmma_f16(a0.v, bu1.v, u[0][1].v);
    u[1][0].v = wmma_f16(a1.v, bu0.v, u[1][0].v);
    u[1][1].v = wmma_f16(a1.v, bu1.v, u[1][1].v);
    __syncthreads();
  }
  const int halfsel = (lane >> 4) * 8;
#pragma unroll
  for (int mi = 0; mi < 2; ++mi)
#pragma unroll
    for (int ni = 0; ni < 2; ++ni) {
      const int ccol = n0 + nw * 32 + ni * 16 + l16;
#pragma unroll
      for (int v2 = 0; v2 < 8; ++v2)
        FF[(size_t)(m0 + mw * 32 + mi * 16 + halfsel + v2) * N + ccol] =
            (_Float16)(gelu_tanh(g[mi][ni].f[v2]) * u[mi][ni].f[v2]);
    }
}

// ---------------------------------------------------------------------------
__global__ __launch_bounds__(256) void laurel_kernel(
    const float* __restrict__ actn, const float* __restrict__ lo,
    const float* __restrict__ plln, float* __restrict__ laurel) {
  __shared__ float sbuf[8];
  const size_t t = blockIdx.x;
  float l8[8];
  float ss = 0.f;
#pragma unroll
  for (int e = 0; e < 8; ++e) {
    l8[e] = lo[t * (size_t)DD + threadIdx.x + 256 * e];
    ss += l8[e] * l8[e];
  }
  ss = block_reduce_sum(ss, sbuf);
  const float r = rsqrtf(ss * (1.0f / DD) + EPSF);
#pragma unroll
  for (int e = 0; e < 8; ++e) {
    const int d = threadIdx.x + 256 * e;
    const size_t idx = t * (size_t)DD + d;
    laurel[idx] = actn[idx] + l8[e] * r * (plln[d] + 1.0f);
  }
}

// ---------------------------------------------------------------------------
__global__ __launch_bounds__(256) void headnorm_rope_kernel(
    const float* __restrict__ lin, const float* __restrict__ cosb,
    const float* __restrict__ sinb, _Float16* __restrict__ dst, int nheads,
    int do_rope) {
  __shared__ float sbuf[8];
  __shared__ float sh[HD];
  const int t = blockIdx.x;  // b*SEQ + s
  const int hh = blockIdx.y;
  const int b = t / SEQ, s = t % SEQ;
  const int d = threadIdx.x;
  const float u = lin[(size_t)t * (nheads * HD) + hh * HD + d];
  const float ssq = block_reduce_sum(u * u, sbuf);
  const float un = u * rsqrtf(ssq * (1.0f / HD) + EPSF);
  float o = un;
  if (do_rope) {
    sh[d] = un;
    __syncthreads();
    const float part = sh[d ^ 128];
    const float cu = cosb[(size_t)s * HD + d];
    const float su = sinb[(size_t)s * HD + d];
    o = (d < 128) ? (un * cu - part * su) : (un * cu + part * su);
  }
  dst[(((size_t)(b * nheads + hh)) * SEQ + s) * HD + d] = (_Float16)o;
}

// ---------------------------------------------------------------------------
// Sliding-window flash attention, S^T formulation: S^T = K * Q^T.
// K tile (row-major in LDS) is the A operand -> contiguous b128 fragment
// reads. Q is the B operand, loaded ONCE per block into registers. K/V tiles
// are DOUBLE-BUFFERED with gfx1250 async global->LDS copies (ASYNCcnt): the
// next 32-key tile streams in while the current one is computed.
// grid (SEQ/64, B*NH), block 128 (4 waves), 16 q-rows per wave.
__global__ __launch_bounds__(128) void flash_attn_kernel(
    const _Float16* __restrict__ Qh, const _Float16* __restrict__ Kh,
    const _Float16* __restrict__ Vh, float* __restrict__ O) {
  __shared__ __align__(16) _Float16 Ks[2][32][HD + 8];
  __shared__ __align__(16) _Float16 Vs[2][32][HD + 8];
  __shared__ __align__(16) _Float16 Ps[4][16][40];
  const int bh = blockIdx.y;
  const int b = bh / NH, h = bh % NH;
  const int kvh = h / (NH / NKV);
  const _Float16* qmem = Qh + (size_t)bh * SEQ * HD;
  const _Float16* kmem = Kh + (size_t)(b * NKV + kvh) * SEQ * HD;
  const _Float16* vmem = Vh + (size_t)(b * NKV + kvh) * SEQ * HD;
  const int qblk = blockIdx.x * 64;
  const int wvid = threadIdx.x >> 5;
  const int lane = threadIdx.x & 31;
  const int q0 = qblk + wvid * 16;
  const int l16 = lane & 15;
  const int halfsel = (lane >> 4) * 8;

  auto stage = [&](int buf, int kt) {
#pragma unroll
    for (int it = 0; it < 8; ++it) {
      const int e = threadIdx.x + it * 128;  // 32 rows * 32 8-half chunks
      const int r = e >> 5, c8 = (e & 31) * 8;
#if USE_ASYNC_LDS
      async_copy16(kmem + (size_t)(kt + r) * HD + c8, &Ks[buf][r][c8]);
      async_copy16(vmem + (size_t)(kt + r) * HD + c8, &Vs[buf][r][c8]);
#else
      *(v8h*)&Ks[buf][r][c8] = *(const v8h*)(kmem + (size_t)(kt + r) * HD + c8);
      *(v8h*)&Vs[buf][r][c8] = *(const v8h*)(vmem + (size_t)(kt + r) * HD + c8);
#endif
    }
  };

  // Q as B-fragments, once: qb[c].h[i] = Q[q0+l16][c*32 + table-order k(i)]
  AFQ qb[8];
#pragma unroll
  for (int c = 0; c < 8; ++c)
#pragma unroll
    for (int i = 0; i < 16; ++i)
      qb[c].h[i] = qmem[(size_t)(q0 + (lane & 15)) * HD + c * 32 + ((lane < 16) ? 0 : 8) + (i & 7) + ((i >> 3) << 4)];
  CFu acc[16] = {};
  float mold = -1e30f;
  float lcur = 0.f;
  const int q_i = q0 + l16;
  int ktlo = qblk - 511;
  if (ktlo < 0) ktlo = 0;
  ktlo &= ~31;
  const int kthi = qblk + 63;

  int buf = 0;
  stage(0, ktlo);
#if USE_ASYNC_LDS
  wait_async0();
#endif
  __syncthreads();

  for (int kt = ktlo; kt <= kthi; kt += 32) {
    if (kt + 32 <= kthi) stage(buf ^ 1, kt + 32);  // prefetch next tile
    CFu st0 = {}, st1 = {};  // S^T tiles: rows = keys (0-15 / 16-31), cols = q
#pragma unroll
    for (int c = 0; c < 8; ++c) {
      const int co = c * 32 + ((lane < 16) ? 0 : 8);
      AFQ a0, a1;
      a0.q[0] = *(const v8h*)&Ks[buf][l16][co];
      a0.q[1] = *(const v8h*)&Ks[buf][l16][co + 16];
      a1.q[0] = *(const v8h*)&Ks[buf][16 + l16][co];
      a1.q[1] = *(const v8h*)&Ks[buf][16 + l16][co + 16];
      st0.v = wmma_f16(a0.v, qb[c].v, st0.v);
      st1.v = wmma_f16(a1.v, qb[c].v, st1.v);
    }
    // mask + column (per-q) online softmax
    float mloc = -1e30f;
#pragma unroll
    for (int v2 = 0; v2 < 8; ++v2) {
      const int j0 = kt + halfsel + v2;
      const int j1 = j0 + 16;
      float a = st0.f[v2] * QSC;
      float c = st1.f[v2] * QSC;
      if (j0 > q_i || q_i - j0 > 511) a = -1e30f;
      if (j1 > q_i || q_i - j1 > 511) c = -1e30f;
      st0.f[v2] = a;
      st1.f[v2] = c;
      mloc = fmaxf(mloc, fmaxf(a, c));
    }
    mloc = fmaxf(mloc, __shfl_xor(mloc, 16, 32));
    const float mn = fmaxf(mold, mloc);
    const float sc = __expf(mold - mn);
    float rs = 0.f;
#pragma unroll
    for (int v2 = 0; v2 < 8; ++v2) {
      const float p0 = __expf(st0.f[v2] - mn);
      const float p1 = __expf(st1.f[v2] - mn);
      rs += p0 + p1;
      Ps[wvid][l16][halfsel + v2] = (_Float16)p0;
      Ps[wvid][l16][16 + halfsel + v2] = (_Float16)p1;
    }
    rs += __shfl_xor(rs, 16, 32);
    lcur = lcur * sc + rs;
    mold = mn;
    // rescale accumulator rows (row = q = v2 + halfsel)
    float scv[8];
#pragma unroll
    for (int v2 = 0; v2 < 8; ++v2) scv[v2] = __shfl(sc, halfsel + v2, 32);
#pragma unroll
    for (int nc = 0; nc < 16; ++nc)
#pragma unroll
      for (int v2 = 0; v2 < 8; ++v2) acc[nc].f[v2] *= scv[v2];
    // P as A-fragment (rows = q, k = key 0..31)
    AFQ pf;
    {
      const int kbp = (lane < 16) ? 0 : 8;
      pf.q[0] = *(const v8h*)&Ps[wvid][l16][kbp];
      pf.q[1] = *(const v8h*)&Ps[wvid][l16][kbp + 16];
    }
#pragma unroll
    for (int nc = 0; nc < 16; ++nc) {
      AFQ vf;
#pragma unroll
      for (int s = 0; s < 16; ++s)
        vf.h[s] = Vs[buf][((lane < 16) ? 0 : 8) + (s & 7) + ((s >> 3) << 4)][nc * 16 + l16];
      acc[nc].v = wmma_f16(pf.v, vf.v, acc[nc].v);
    }
#if USE_ASYNC_LDS
    wait_async0();  // next tile has landed
#endif
    __syncthreads();  // all waves done with current tile + see next tile
    buf ^= 1;
  }
  const float linv = 1.0f / lcur;
  float invv[8];
#pragma unroll
  for (int v2 = 0; v2 < 8; ++v2) invv[v2] = __shfl(linv, halfsel + v2, 32);
#pragma unroll
  for (int v2 = 0; v2 < 8; ++v2) {
    const int srow = q0 + halfsel + v2;
    float* orow = O + ((size_t)(b * SEQ + srow)) * (NH * HD) + h * HD;
#pragma unroll
    for (int nc = 0; nc < 16; ++nc)
      orow[nc * 16 + l16] = acc[nc].f[v2] * invv[v2];
  }
}

// ---------------------------------------------------------------------------
__global__ __launch_bounds__(256) void attn_post_kernel(
    const float* __restrict__ act, const float* __restrict__ araw,
    const float* __restrict__ laurel, const float* __restrict__ paln,
    const float* __restrict__ pfln, float* __restrict__ al_out,
    float* __restrict__ h_out) {
  __shared__ float sbuf[8];
  const size_t t = blockIdx.x;
  float ar[8];
  float ss = 0.f;
#pragma unroll
  for (int e = 0; e < 8; ++e) {
    ar[e] = araw[t * (size_t)DD + threadIdx.x + 256 * e];
    ss += ar[e] * ar[e];
  }
  ss = block_reduce_sum(ss, sbuf);
  const float r1 = rsqrtf(ss * (1.0f / DD) + EPSF);
  float al[8];
  float ss2 = 0.f;
#pragma unroll
  for (int e = 0; e < 8; ++e) {
    const int d = threadIdx.x + 256 * e;
    const size_t idx = t * (size_t)DD + d;
    const float attn = ar[e] * r1 * (paln[d] + 1.0f);
    al[e] = (act[idx] + attn + laurel[idx]) * INV_SQRT2;
    al_out[idx] = al[e];
    ss2 += al[e] * al[e];
  }
  ss2 = block_reduce_sum(ss2, sbuf);
  const float r2 = rsqrtf(ss2 * (1.0f / DD) + EPSF);
#pragma unroll
  for (int e = 0; e < 8; ++e) {
    const int d = threadIdx.x + 256 * e;
    h_out[t * (size_t)DD + d] = al[e] * r2 * (pfln[d] + 1.0f);
  }
}

// ---------------------------------------------------------------------------
__global__ __launch_bounds__(256) void ff_post_kernel(
    const float* __restrict__ al, const float* __restrict__ ffo,
    const float* __restrict__ pffln, float* __restrict__ activated) {
  __shared__ float sbuf[8];
  const size_t t = blockIdx.x;
  float fo[8];
  float ss = 0.f;
#pragma unroll
  for (int e = 0; e < 8; ++e) {
    fo[e] = ffo[t * (size_t)DD + threadIdx.x + 256 * e];
    ss += fo[e] * fo[e];
  }
  ss = block_reduce_sum(ss, sbuf);
  const float r = rsqrtf(ss * (1.0f / DD) + EPSF);
#pragma unroll
  for (int e = 0; e < 8; ++e) {
    const int d = threadIdx.x + 256 * e;
    const size_t idx = t * (size_t)DD + d;
    activated[idx] = al[idx] + fo[e] * r * (pffln[d] + 1.0f);
  }
}

// ---------------------------------------------------------------------------
__global__ __launch_bounds__(256) void final_correct_kernel(
    const float* __restrict__ X, const float* __restrict__ coefs,
    const float* __restrict__ m2, const float* __restrict__ ccw,
    const float* __restrict__ act, const float* __restrict__ activated,
    const float* __restrict__ cscale, float* __restrict__ out) {
  const size_t t = blockIdx.x;
  float m4[NIM];
#pragma unroll
  for (int k = 0; k < NIM; ++k) m4[k] = m2[t * NIM + k];
  float cc[NIM];
#pragma unroll
  for (int i = 0; i < NIM; ++i) {
    float c = 0.f;
#pragma unroll
    for (int k = 0; k < NIM; ++k) c += m4[k] * clipw(ccw[i * NIM + k]);
    cc[i] = c + 1.0f;
  }
  float cf[NIM * NIM];
#pragma unroll
  for (int e2 = 0; e2 < NIM * NIM; ++e2) cf[e2] = coefs[t * 16 + e2];
  const size_t plane = (size_t)BS_ * DD;
#pragma unroll
  for (int e = 0; e < 8; ++e) {
    const int d = threadIdx.x + 256 * e;
    const size_t idx = t * (size_t)DD + d;
    float xv[NIM];
#pragma unroll
    for (int j = 0; j < NIM; ++j) xv[j] = X[(size_t)j * plane + idx];
    const float innov = activated[idx] - act[idx];
#pragma unroll
    for (int i = 0; i < NIM; ++i) {
      float pred = xv[i];
#pragma unroll
      for (int j = 0; j < NIM; ++j) pred += cf[i * NIM + j] * xv[j];
      float o = pred + cc[i] * innov;
      if (i == 0) o *= cscale[d];
      out[(size_t)i * plane + idx] = o;
    }
  }
}

// ---------------------------------------------------------------------------
extern "C" void kernel_launch(void* const* d_in, const int* in_sizes, int n_in,
                              void* d_out, int out_size, void* d_ws,
                              size_t ws_size, hipStream_t stream) {
  (void)in_sizes; (void)n_in; (void)out_size; (void)ws_size;
  const float* X    = (const float*)d_in[0];
  const float* COSB = (const float*)d_in[1];
  const float* SINB = (const float*)d_in[2];
  const float* RNW  = (const float*)d_in[3];
  const float* WM   = (const float*)d_in[4];
  const float* PCW  = (const float*)d_in[5];
  const float* CCW  = (const float*)d_in[6];
  const float* CSC  = (const float*)d_in[7];
  const float* ILN  = (const float*)d_in[8];
  const float* PALN = (const float*)d_in[9];
  const float* PFLN = (const float*)d_in[10];
  const float* PFFLN= (const float*)d_in[11];
  const float* PLLN = (const float*)d_in[12];
  const float* WQ   = (const float*)d_in[13];
  const float* WK   = (const float*)d_in[14];
  const float* WV   = (const float*)d_in[15];
  const float* WO   = (const float*)d_in[16];
  const float* WLL  = (const float*)d_in[17];
  const float* WLR  = (const float*)d_in[18];
  const float* WGATE= (const float*)d_in[19];
  const float* WUP  = (const float*)d_in[20];
  const float* WDOWN= (const float*)d_in[21];

  float* WS = (float*)d_ws;
  const size_t ROW = (size_t)BS_ * DD;  // 8M floats
  size_t off = 0;
  float* mbuf = WS + off;      off += 16384;
  float* m2buf = WS + off;     off += 16384;
  float* coefsbuf = WS + off;  off += 65536 + 32768;  // pad to 131072 total
  float* act = WS + off;       off += ROW;
  float* actn = WS + off;      off += ROW;   // reused: O, then h
  float* qlin = WS + off;      off += ROW;   // reused: attn_raw, then ffo
  float* klin = WS + off;      off += (size_t)BS_ * 512;
  float* vlin = WS + off;      off += (size_t)BS_ * 512;
  float* lo1 = WS + off;       off += (size_t)BS_ * 64;
  float* lo = WS + off;        off += ROW;   // reused: attn_laurel
  float* laurel = WS + off;    off += ROW;   // reused: activated
  _Float16* hb = (_Float16*)(WS + off);
  size_t hoff = 0;
  _Float16* qf16 = hb + hoff;  hoff += (size_t)BB * NH * SEQ * HD;
  _Float16* kf16 = hb + hoff;  hoff += (size_t)BB * NKV * SEQ * HD;
  _Float16* vf16 = hb + hoff;  hoff += (size_t)BB * NKV * SEQ * HD;
  _Float16* ffh = hb + hoff;   hoff += (size_t)BS_ * FFDIM;

  float* Obuf = actn;
  float* attn_raw = qlin;
  float* attn_laurel = lo;
  float* hbuf = actn;
  float* ffo = qlin;
  float* activated = laurel;

  modalities_kernel<<<BS_, 256, 0, stream>>>(X, RNW, WM, mbuf);
  coefs_act_kernel<<<BS_, 256, 0, stream>>>(X, mbuf, PCW, ILN, coefsbuf, act,
                                            actn);
  gemm_nt_kernel<float><<<dim3(BS_ / 128, 2048 / 64), 256, 0, stream>>>(
      actn, WQ, qlin, BS_, 2048, 2048);
  gemm_nt_kernel<float><<<dim3(BS_ / 128, 512 / 64), 256, 0, stream>>>(
      actn, WK, klin, BS_, 512, 2048);
  gemm_nt_kernel<float><<<dim3(BS_ / 128, 512 / 64), 256, 0, stream>>>(
      actn, WV, vlin, BS_, 512, 2048);
  gemm_nt_kernel<float><<<dim3(BS_ / 128, 1), 256, 0, stream>>>(
      actn, WLL, lo1, BS_, 64, 2048);
  gemm_nt_kernel<float><<<dim3(BS_ / 128, 2048 / 64), 256, 0, stream>>>(
      lo1, WLR, lo, BS_, 2048, 64);
  laurel_kernel<<<BS_, 256, 0, stream>>>(actn, lo, PLLN, laurel);
  headnorm_rope_kernel<<<dim3(BS_, NH), 256, 0, stream>>>(qlin, COSB, SINB,
                                                          qf16, NH, 1);
  headnorm_rope_kernel<<<dim3(BS_, NKV), 256, 0, stream>>>(klin, COSB, SINB,
                                                           kf16, NKV, 1);
  headnorm_rope_kernel<<<dim3(BS_, NKV), 256, 0, stream>>>(vlin, COSB, SINB,
                                                           vf16, NKV, 0);
  flash_attn_kernel<<<dim3(SEQ / 64, BB * NH), 128, 0, stream>>>(qf16, kf16,
                                                                 vf16, Obuf);
  gemm_nt_kernel<float><<<dim3(BS_ / 128, 2048 / 64), 256, 0, stream>>>(
      Obuf, WO, attn_raw, BS_, 2048, 2048);
  attn_post_kernel<<<BS_, 256, 0, stream>>>(act, attn_raw, laurel, PALN, PFLN,
                                            attn_laurel, hbuf);
  gemm_gate_up_kernel<<<dim3(BS_ / 128, FFDIM / 64), 256, 0, stream>>>(
      hbuf, WGATE, WUP, ffh, BS_, FFDIM, 2048);
  gemm_nt_kernel<_Float16><<<dim3(BS_ / 128, 2048 / 64), 256, 0, stream>>>(
      ffh, WDOWN, ffo, BS_, 2048, FFDIM);
  ff_post_kernel<<<BS_, 256, 0, stream>>>(attn_laurel, ffo, PFFLN, activated);
  modalities_kernel<<<BS_, 256, 0, stream>>>(activated, RNW, WM, m2buf);
  final_correct_kernel<<<BS_, 256, 0, stream>>>(X, coefsbuf, m2buf, CCW, act,
                                                activated, CSC, (float*)d_out);
}
